// SAGE_15187004359143
// MI455X (gfx1250) — matrix operand order repaired and checked by reading, not compiled, not run on hardware
//
#include <hip/hip_runtime.h>

typedef __attribute__((ext_vector_type(16))) __bf16 v16bf;
typedef __attribute__((ext_vector_type(8)))  float  v8f;

#define N_NODES 100000
#define N_EDGES 1600000
#define FEAT    128

// ---------------- utility kernels ----------------

__global__ void zero_f32(float* __restrict__ p, unsigned n) {
  unsigned t = blockIdx.x * blockDim.x + threadIdx.x;
  if (t < n) p[t] = 0.0f;
}

__global__ void deg_count(const int* __restrict__ dst, float* __restrict__ deg, int nE) {
  int t = blockIdx.x * blockDim.x + threadIdx.x;
  if (t < nE)
    __hip_atomic_fetch_add(&deg[dst[t]], 1.0f, __ATOMIC_RELAXED, __HIP_MEMORY_SCOPE_AGENT);
}

__global__ void invert_deg(float* __restrict__ deg, int n) {
  int t = blockIdx.x * blockDim.x + threadIdx.x;
  if (t < n) deg[t] = 1.0f / fmaxf(deg[t], 1.0f);
}

// One thread per (edge, feature). 128 consecutive threads read one contiguous
// src row (512B) and atomically add into one contiguous dst row. h fits in the
// 192MB L2, so gathers and fp32 atomics resolve at L2 bandwidth.
__global__ void scatter128(const float* __restrict__ h, const int* __restrict__ src,
                           const int* __restrict__ dst, float* __restrict__ agg, int nE) {
  unsigned t = blockIdx.x * blockDim.x + threadIdx.x;
  unsigned e = t >> 7;
  unsigned f = t & 127u;
  if (e >= (unsigned)nE) return;
  float v = h[(size_t)src[e] * FEAT + f];
  __hip_atomic_fetch_add(&agg[(size_t)dst[e] * FEAT + f], v,
                         __ATOMIC_RELAXED, __HIP_MEMORY_SCOPE_AGENT);
}

// fp32 [K][Nout] -> bf16 transposed [Nout][K] so B fragments are contiguous.
__global__ void convert_w_t(const float* __restrict__ W, __bf16* __restrict__ Wt,
                            int K, int Nout) {
  int t = blockIdx.x * blockDim.x + threadIdx.x;
  if (t >= K * Nout) return;
  int k = t / Nout, n = t - k * Nout;
  Wt[(size_t)n * K + k] = (__bf16)W[t];
}

// ---------------- WMMA GEMM ----------------
// A fragment k order this code uses for V_WMMA_*_16X16X32_BF16 (wave32; B matches):
//   lane l: row = l%16, half = l/16
//   element e (0..15): K = k0 + half*16 + e
// -> two contiguous groups of 8 floats, each 32B aligned.
__device__ __forceinline__ v16bf load_a_frag(const float* __restrict__ row,
                                             int k0, int half, float s) {
  const float4* p0 = (const float4*)(row + k0 + half * 16);
  const float4* p1 = (const float4*)(row + k0 + half * 16 + 8);
  float4 u0 = p0[0], u1 = p0[1];
  float4 w0 = p1[0], w1 = p1[1];
  v16bf a;
  a[0]  = (__bf16)(u0.x * s); a[1]  = (__bf16)(u0.y * s);
  a[2]  = (__bf16)(u0.z * s); a[3]  = (__bf16)(u0.w * s);
  a[4]  = (__bf16)(u1.x * s); a[5]  = (__bf16)(u1.y * s);
  a[6]  = (__bf16)(u1.z * s); a[7]  = (__bf16)(u1.w * s);
  a[8]  = (__bf16)(w0.x * s); a[9]  = (__bf16)(w0.y * s);
  a[10] = (__bf16)(w0.z * s); a[11] = (__bf16)(w0.w * s);
  a[12] = (__bf16)(w1.x * s); a[13] = (__bf16)(w1.y * s);
  a[14] = (__bf16)(w1.z * s); a[15] = (__bf16)(w1.w * s);
  return a;
}

// Out[v,:] = H[v,:]@Ws + (AGG[v,:]*invdeg[v])@Wn + bias  (optional relu)
// One wave per 16-row M tile; K = 128 (4 k-blocks of 32); two WMMAs per
// k-block (self + neigh) accumulating into one f32 16x16 tile.
__global__ void sage_wmma_gemm(const float* __restrict__ H,
                               const float* __restrict__ AGG,
                               const float* __restrict__ invdeg,
                               const __bf16* __restrict__ WsT,   // [Nout][K] bf16
                               const __bf16* __restrict__ WnT,   // [Nout][K] bf16
                               const float* __restrict__ bias,   // [Nout]
                               float* __restrict__ Out,          // [M][Nout]
                               int Mtiles, int Nout, int relu) {
  const int K = FEAT;
  int wave = threadIdx.x >> 5;
  int lane = threadIdx.x & 31;
  int mt = blockIdx.x * (blockDim.x >> 5) + wave;
  if (mt >= Mtiles) return;              // wave-uniform: EXEC all-1s past here

  int half = lane >> 4;
  int rsub = lane & 15;
  int rowA = mt * 16 + rsub;             // this lane's A-matrix row
  float sc = invdeg[rowA];
  const float* hrow = H   + (size_t)rowA * K;
  const float* arow = AGG + (size_t)rowA * K;

  v16bf aS[4], aN[4];
#pragma unroll
  for (int kb = 0; kb < 4; ++kb) {
    aS[kb] = load_a_frag(hrow, kb * 32, half, 1.0f);
    aN[kb] = load_a_frag(arow, kb * 32, half, sc);
  }

  for (int n0 = 0; n0 < Nout; n0 += 16) {
    int col = n0 + rsub;                 // B fragment: N = lane%16, K = half*16 + e
    const __bf16* ps = WsT + (size_t)col * K + half * 16;
    const __bf16* pn = WnT + (size_t)col * K + half * 16;
    v8f c = {0.f, 0.f, 0.f, 0.f, 0.f, 0.f, 0.f, 0.f};
#pragma unroll
    for (int kb = 0; kb < 4; ++kb) {
      v16bf bS = *(const v16bf*)(ps + kb * 32);
      c = __builtin_amdgcn_wmma_f32_16x16x32_bf16(false, aS[kb], false, bS,
                                                  (short)0, c, false, false);
      v16bf bN = *(const v16bf*)(pn + kb * 32);
      c = __builtin_amdgcn_wmma_f32_16x16x32_bf16(false, aN[kb], false, bN,
                                                  (short)0, c, false, false);
    }
    // D layout: element r -> row = mt*16 + half*8 + r, col = n0 + rsub
    float bv = bias[col];
#pragma unroll
    for (int r = 0; r < 8; ++r) {
      float v = c[r] + bv;
      if (relu) v = fmaxf(v, 0.0f);
      Out[(size_t)(mt * 16 + half * 8 + r) * Nout + col] = v;
    }
  }
}

// ---------------- launch ----------------

extern "C" void kernel_launch(void* const* d_in, const int* in_sizes, int n_in,
                              void* d_out, int out_size, void* d_ws, size_t ws_size,
                              hipStream_t stream) {
  const float* x    = (const float*)d_in[0];
  const int*   src  = (const int*)d_in[1];
  const int*   dstp = (const int*)d_in[2];
  const float* W[6] = { (const float*)d_in[3], (const float*)d_in[4],    // Ws1, Wn1
                        (const float*)d_in[6], (const float*)d_in[7],    // Ws2, Wn2
                        (const float*)d_in[9], (const float*)d_in[10] }; // Ws3, Wn3
  const float* b1 = (const float*)d_in[5];
  const float* b2 = (const float*)d_in[8];
  const float* b3 = (const float*)d_in[11];

  char* ws = (char*)d_ws;
  size_t off = 0;
  auto carve = [&](size_t bytes) -> void* {
    void* p = ws + off;
    off += (bytes + 255) & ~(size_t)255;
    return p;
  };
  float* agg    = (float*)carve((size_t)N_NODES * FEAT * sizeof(float));
  float* invdeg = (float*)carve((size_t)N_NODES * sizeof(float));
  float* h1     = (float*)carve((size_t)N_NODES * FEAT * sizeof(float));
  float* h2     = (float*)carve((size_t)N_NODES * FEAT * sizeof(float));
  const int wK[6] = {128, 128, 128, 128, 128, 128};
  const int wN[6] = {128, 128, 128, 128, 64, 64};
  __bf16* Wt[6];
  for (int i = 0; i < 6; ++i)
    Wt[i] = (__bf16*)carve((size_t)wK[i] * wN[i] * sizeof(__bf16));

  // degree -> inverse degree (same for all layers; depends only on dst)
  zero_f32<<<(N_NODES + 255) / 256, 256, 0, stream>>>(invdeg, N_NODES);
  deg_count<<<(N_EDGES + 255) / 256, 256, 0, stream>>>(dstp, invdeg, N_EDGES);
  invert_deg<<<(N_NODES + 255) / 256, 256, 0, stream>>>(invdeg, N_NODES);

  // weights -> bf16, transposed
  for (int i = 0; i < 6; ++i)
    convert_w_t<<<(wK[i] * wN[i] + 255) / 256, 256, 0, stream>>>(W[i], Wt[i], wK[i], wN[i]);

  const unsigned zgrid  = (N_NODES * FEAT) / 256;              // 50000
  const unsigned sgrid  = ((unsigned)N_EDGES * FEAT) / 256;    // 800000
  const int      Mtiles = N_NODES / 16;                        // 6250 (exact)
  const int      ggrid  = (Mtiles + 7) / 8;                    // 8 waves/block

  // layer 1: x -> h1 (relu)
  zero_f32<<<zgrid, 256, 0, stream>>>(agg, (unsigned)N_NODES * FEAT);
  scatter128<<<sgrid, 256, 0, stream>>>(x, src, dstp, agg, N_EDGES);
  sage_wmma_gemm<<<ggrid, 256, 0, stream>>>(x, agg, invdeg, Wt[0], Wt[1], b1,
                                            h1, Mtiles, 128, 1);
  // layer 2: h1 -> h2 (relu)
  zero_f32<<<zgrid, 256, 0, stream>>>(agg, (unsigned)N_NODES * FEAT);
  scatter128<<<sgrid, 256, 0, stream>>>(h1, src, dstp, agg, N_EDGES);
  sage_wmma_gemm<<<ggrid, 256, 0, stream>>>(h1, agg, invdeg, Wt[2], Wt[3], b2,
                                            h2, Mtiles, 128, 1);
  // layer 3: h2 -> out (no relu, Nout = 64)
  zero_f32<<<zgrid, 256, 0, stream>>>(agg, (unsigned)N_NODES * FEAT);
  scatter128<<<sgrid, 256, 0, stream>>>(h2, src, dstp, agg, N_EDGES);
  sage_wmma_gemm<<<ggrid, 256, 0, stream>>>(h2, agg, invdeg, Wt[4], Wt[5], b3,
                                            (float*)d_out, Mtiles, 64, 0);

  (void)in_sizes; (void)n_in; (void)out_size; (void)ws_size;
}
